// baseline_lstm_61753039782391
// MI455X (gfx1250) — matrix operand order repaired and checked by reading, not compiled, not run on hardware
//
#include <hip/hip_runtime.h>

#define IN_F 60
#define ENC  32
#define HH   64
#define BB   256
#define TT   4096
#define NG   256   // 4*H gate columns
// A K layout: [0..59]=c_t, [60..63]=zero pad, [64..127]=h

typedef _Float16 v16h __attribute__((ext_vector_type(16)));
typedef _Float16 v8h  __attribute__((ext_vector_type(8)));
typedef float    v8f  __attribute__((ext_vector_type(8)));
typedef float    f4   __attribute__((ext_vector_type(4)));

union AV { v16h v; v8h h[2]; };
union BV { v16h v; _Float16 e[16]; };

__device__ __forceinline__ float fast_rcp(float x) {
#if __has_builtin(__builtin_amdgcn_rcpf)
  return __builtin_amdgcn_rcpf(x);
#else
  return 1.0f / x;
#endif
}

__device__ __forceinline__ float fast_tanh(float x) {
#if __has_builtin(__builtin_amdgcn_tanhf)
  return __builtin_amdgcn_tanhf(x);   // v_tanh_f32 (1 trans op)
#else
  float t = __builtin_exp2f(x * 2.8853900817779268f);  // exp(2x)
  return 1.0f - 2.0f * fast_rcp(t + 1.0f);
#endif
}

__device__ __forceinline__ float fast_sig(float x) {
  return 0.5f * fast_tanh(0.5f * x) + 0.5f;
}

__device__ __forceinline__ unsigned pack2h(float a, float b) {
  _Float16 ha = (_Float16)a, hb = (_Float16)b;
  unsigned short ua = __builtin_bit_cast(unsigned short, ha);
  unsigned short ub = __builtin_bit_cast(unsigned short, hb);
  return (unsigned)ua | ((unsigned)ub << 16);
}

__global__ __launch_bounds__(512)
void lstm_persistent_kernel(const float* __restrict__ c,
                            const float* __restrict__ W_enc,
                            const float* __restrict__ b_enc,
                            const float* __restrict__ W_ih_f,
                            const float* __restrict__ W_hh_f,
                            const float* __restrict__ b_f,
                            const float* __restrict__ W_ih_b,
                            const float* __restrict__ W_hh_b,
                            const float* __restrict__ b_b,
                            float* __restrict__ hidden)
{
  // A staging: 16 rows x 128 K (f16), row stride 136 (272B, 16B aligned,
  // 4r-bank rotation -> conflict-free b128 loads)
  __shared__ alignas(16) _Float16 astage[16][136];
  // gates, column-major [n][m], stride 20 f32 (80B) -> conflict-free f4 stores
  __shared__ alignas(16) float gbuf[NG][20];

  const int tid  = threadIdx.x;
  const int lane = tid & 31;
  const int wave = tid >> 5;
  const int grp  = blockIdx.x & 15;   // batch group (16 rows)
  const int dir  = blockIdx.x >> 4;   // 0 = fwd, 1 = bwd
  const int b0   = grp * 16;

  const float* __restrict__ Wih = dir ? W_ih_b : W_ih_f;
  const float* __restrict__ Whh = dir ? W_hh_b : W_hh_f;
  const float* __restrict__ bg  = dir ? b_b    : b_f;

  // ---------------- one-time weight gather into registers ----------------
  // B packing used here (contiguous k, same order as A): lane<16 -> col=n0+lane, VGPR v
  // holds K=2v,2v+1 ; lane>=16 -> col=n0+lane-16, VGPR v holds K=16+2v,17+2v.
  const int n0  = wave * 16;
  const int col = n0 + (lane & 15);
  const int kh  = (lane < 16) ? 0 : 16;

  float wih[ENC];
  #pragma unroll
  for (int e = 0; e < ENC; ++e) wih[e] = Wih[col * ENC + e];

  BV bop[4];
  #pragma unroll
  for (int q = 0; q < 4; ++q) {
    #pragma unroll
    for (int v = 0; v < 8; ++v) {
      #pragma unroll
      for (int hx = 0; hx < 2; ++hx) {
        int K = 32 * q + kh + 2 * v + hx;
        float w;
        if (K < 64) {
          w = 0.0f;
          if (K < IN_F) {            // W_comb = W_ih @ W_enc  (encoder folded)
            #pragma unroll
            for (int e = 0; e < ENC; ++e) w += wih[e] * W_enc[e * IN_F + K];
          }
        } else {
          w = Whh[col * HH + (K - 64)];
        }
        bop[q].e[2 * v + hx] = (_Float16)w;
      }
    }
  }
  // bias with encoder bias folded through W_ih
  float bn = bg[col];
  #pragma unroll
  for (int e = 0; e < ENC; ++e) bn += wih[e] * b_enc[e];

  // elementwise mapping: wave = row, lane -> h cols {lane, lane+32}
  const int rew = wave;
  float cs0 = 0.0f, cs1 = 0.0f;

  // zero h region + K=60..63 pad of astage
  *(unsigned*)&astage[tid >> 5][64 + (lane << 1)] = 0u;
  if (tid < 32) *(unsigned*)&astage[tid >> 1][60 + ((tid & 1) << 1)] = 0u;

  // x loader mapping: 480 threads, 16 rows x 30 float2 segments (60 f32/row)
  const int  lrow = tid / 30, lseg = tid % 30;
  const bool is_loader = (tid < 480);
  const float* __restrict__ crow =
      c + (size_t)(b0 + (is_loader ? lrow : 0)) * (size_t)TT * IN_F;
  const float* __restrict__ prow =
      c + (size_t)(b0 + (tid & 15)) * (size_t)TT * IN_F;

  // prologue: stage x for t=0
  if (is_loader) {
    int t_x = dir ? (TT - 1) : 0;
    float2 x2 = *(const float2*)(crow + (size_t)t_x * IN_F + lseg * 2);
    *(unsigned*)&astage[lrow][lseg * 2] = pack2h(x2.x, x2.y);
  }
  __syncthreads();

  const int arow = lane & 15;
  const int aoff = (lane < 16) ? 0 : 16;

  for (int t = 0; t < TT; ++t) {
    // -------- phase A: A operands from LDS, WMMA, stage gates --------
    AV a[4];
    #pragma unroll
    for (int q = 0; q < 4; ++q) {
      a[q].h[0] = *(const v8h*)&astage[arow][32 * q + aoff];
      a[q].h[1] = *(const v8h*)&astage[arow][32 * q + aoff + 8];
    }

    // kick off next-step x load early (latency hidden behind WMMA/barrier)
    float2 xnext;
    const bool doload = is_loader && (t + 1 < TT);
    if (doload) {
      int t_x = dir ? (TT - 2 - t) : (t + 1);
      xnext = *(const float2*)(crow + (size_t)t_x * IN_F + lseg * 2);
    }
    if (tid < 16 && t + 8 < TT) {
      int t_pf = dir ? (TT - 9 - t) : (t + 8);
      __builtin_prefetch(prow + (size_t)t_pf * IN_F, 0, 1); // global_prefetch_b8
    }

    v8f acc;
    #pragma unroll
    for (int r = 0; r < 8; ++r) acc[r] = bn;   // bias pre-loaded into C
    acc = __builtin_amdgcn_wmma_f32_16x16x32_f16(false, a[0].v, false, bop[0].v,
                                                 (short)0, acc, false, false);
    acc = __builtin_amdgcn_wmma_f32_16x16x32_f16(false, a[1].v, false, bop[1].v,
                                                 (short)0, acc, false, false);
    acc = __builtin_amdgcn_wmma_f32_16x16x32_f16(false, a[2].v, false, bop[2].v,
                                                 (short)0, acc, false, false);
    acc = __builtin_amdgcn_wmma_f32_16x16x32_f16(false, a[3].v, false, bop[3].v,
                                                 (short)0, acc, false, false);

    // D tile -> gates LDS (col-major). lane<16: M=0..7 in acc[0..7];
    // lane>=16: M=8..15.
    {
      int n  = n0 + (lane & 15);
      int m0 = (lane < 16) ? 0 : 8;
      f4 lo = {acc[0], acc[1], acc[2], acc[3]};
      f4 hi = {acc[4], acc[5], acc[6], acc[7]};
      *(f4*)&gbuf[n][m0]     = lo;
      *(f4*)&gbuf[n][m0 + 4] = hi;
    }
    __syncthreads();

    // -------- phase B: commit next x, elementwise cell update --------
    if (doload)
      *(unsigned*)&astage[lrow][lseg * 2] = pack2h(xnext.x, xnext.y);

    {
      int j  = lane;         // h col 0..31
      int j2 = lane + 32;    // h col 32..63
      float gi0 = gbuf[j][rew],        gi1 = gbuf[j2][rew];
      float gf0 = gbuf[64 + j][rew],   gf1 = gbuf[64 + j2][rew];
      float gg0 = gbuf[128 + j][rew],  gg1 = gbuf[128 + j2][rew];
      float go0 = gbuf[192 + j][rew],  go1 = gbuf[192 + j2][rew];
      cs0 = fast_sig(gf0) * cs0 + fast_sig(gi0) * fast_tanh(gg0);
      cs1 = fast_sig(gf1) * cs1 + fast_sig(gi1) * fast_tanh(gg1);
      _Float16 h0 = (_Float16)(fast_sig(go0) * fast_tanh(cs0));
      _Float16 h1 = (_Float16)(fast_sig(go1) * fast_tanh(cs1));
      astage[rew][64 + j]  = h0;
      astage[rew][64 + j2] = h1;
    }
    __syncthreads();
  }

  // final CELL states (reference feeds c_n into the final FC)
  hidden[(size_t)(b0 + rew) * 128 + dir * 64 + lane]      = cs0;
  hidden[(size_t)(b0 + rew) * 128 + dir * 64 + 32 + lane] = cs1;
}

__global__ __launch_bounds__(256)
void final_fc_kernel(const float* __restrict__ hidden,
                     const float* __restrict__ W_fin,
                     const float* __restrict__ b_fin,
                     float* __restrict__ out)
{
  int b = threadIdx.x;
  float a0 = b_fin[0], a1 = b_fin[1], a2 = b_fin[2];
  #pragma unroll 8
  for (int k = 0; k < 128; ++k) {
    float h = hidden[b * 128 + k];
    a0 += h * W_fin[k];
    a1 += h * W_fin[128 + k];
    a2 += h * W_fin[256 + k];
  }
  out[b * 3 + 0] = a0;
  out[b * 3 + 1] = a1;
  out[b * 3 + 2] = a2;
}

extern "C" void kernel_launch(void* const* d_in, const int* in_sizes, int n_in,
                              void* d_out, int out_size, void* d_ws, size_t ws_size,
                              hipStream_t stream)
{
  (void)in_sizes; (void)n_in; (void)out_size; (void)ws_size;
  const float* c      = (const float*)d_in[0];
  const float* W_enc  = (const float*)d_in[1];
  const float* b_enc  = (const float*)d_in[2];
  const float* W_ih_f = (const float*)d_in[3];
  const float* W_hh_f = (const float*)d_in[4];
  const float* b_f    = (const float*)d_in[5];
  const float* W_ih_b = (const float*)d_in[6];
  const float* W_hh_b = (const float*)d_in[7];
  const float* b_b    = (const float*)d_in[8];
  const float* W_fin  = (const float*)d_in[9];
  const float* b_fin  = (const float*)d_in[10];

  float* hidden = (float*)d_ws;   // [256][128] f32 = 128 KB

  lstm_persistent_kernel<<<32, 512, 0, stream>>>(
      c, W_enc, b_enc, W_ih_f, W_hh_f, b_f, W_ih_b, W_hh_b, b_b, hidden);
  final_fc_kernel<<<1, 256, 0, stream>>>(hidden, W_fin, b_fin, (float*)d_out);
}
